// SimpleGNN_19396072309227
// MI455X (gfx1250) — matrix operand order repaired and checked by reading, compiled, not run on hardware
//
#include <hip/hip_runtime.h>

// ---------------------------------------------------------------------------
// CDNA5 (gfx1250, wave32) GNN: all dense math mapped to v_wmma_f32_16x16x32_f16
// x feature tensor stored f16 (all consumers feed WMMA f16 anyway) -> halves
// L2 gather bytes in the dominant edge kernel and removes cvt work.
// agg stays f32 (global_atomic_add_f32 segment-sum).
// ---------------------------------------------------------------------------

typedef __attribute__((ext_vector_type(16))) _Float16 v16h;
typedef __attribute__((ext_vector_type(8)))  _Float16 v8h;
typedef __attribute__((ext_vector_type(8)))  float    v8f;

#define WPB 8              // waves per block
#define TPB (WPB * 32)

__device__ __forceinline__ v8f wmma16(v16h a, v16h b, v8f c) {
  // D = A(16x32 f16) x B(32x16 f16) + C(16x16 f32)
  return __builtin_amdgcn_wmma_f32_16x16x32_f16(false, a, false, b, (short)0, c,
                                                false, false);
}

__device__ __forceinline__ v8f splat8(float v) {
  v8f c;
  #pragma unroll
  for (int r = 0; r < 8; ++r) c[r] = v;
  return c;
}

// A fragment (16x32, f16) from a 32-float (f32) row owned by lane's M index.
// ISA layout (16-bit A 16x32): lanes 0-15 hold K 0..7 (h=0..7) and K 16..23
// (h=8..15); lanes 16-31 hold K 8..15 and K 24..31.
__device__ __forceinline__ v16h a_frag_from_row_f32(const float4* row4, int lane) {
  const int q = (lane & 16) ? 2 : 0;
  float4 s0 = row4[q];
  float4 s1 = row4[q + 1];
  float4 s2 = row4[q + 4];
  float4 s3 = row4[q + 5];
  float f[16] = { s0.x, s0.y, s0.z, s0.w, s1.x, s1.y, s1.z, s1.w,
                  s2.x, s2.y, s2.z, s2.w, s3.x, s3.y, s3.z, s3.w };
  v16h a;
  #pragma unroll
  for (int h = 0; h < 16; ++h) a[h] = (_Float16)f[h];
  return a;
}

// A fragment from a 32-half (f16) row: two contiguous 16-byte segments
// per lane -> two global_load_b128, no conversions.
__device__ __forceinline__ v16h a_frag_from_row_f16(const _Float16* row, int lane) {
  const int kb = (lane & 16) ? 8 : 0;
  v8h s0 = *(const v8h*)(row + kb);        // K kb..kb+7
  v8h s1 = *(const v8h*)(row + kb + 16);   // K kb+16..kb+23
  v16h a;
  #pragma unroll
  for (int h = 0; h < 8; ++h) { a[h] = s0[h]; a[8 + h] = s1[h]; }
  return a;
}

// B fragment (32x16, f16) from f32 weight matrix W (K x 32 row-major),
// rows [k0,k0+32) cols [n0,n0+16).  Lanes 0-15 hold K 0..7,16..23, lanes 16-31
// hold K 8..15,24..31.  Rows >= kRows are zero pad.
__device__ __forceinline__ v16h b_frag_from_w(const float* W, int kRows,
                                              int k0, int n0, int lane) {
  const int n  = n0 + (lane & 15);
  const int kb = k0 + ((lane & 16) ? 8 : 0);
  v16h b;
  #pragma unroll
  for (int h = 0; h < 16; ++h) {
    const int k = kb + (h < 8 ? h : h + 8);
    const float w = (k < kRows) ? W[k * 32 + n] : 0.0f;
    b[h] = (_Float16)w;
  }
  return b;
}

// ReLU(D) -> LDS 16x32 f32 tile (row-major), for D->A re-layout.
__device__ __forceinline__ void relu_store_lds(float* lds, v8f d, int n0, int lane) {
  const int n  = n0 + (lane & 15);
  const int mb = (lane & 16) ? 8 : 0;
  #pragma unroll
  for (int r = 0; r < 8; ++r)
    lds[(mb + r) * 32 + n] = fmaxf(d[r], 0.0f);
}

// A fragment from an LDS f32 row (ds_load_b128 path).
__device__ __forceinline__ v16h a_frag_from_lds(const float* ldsrow, int lane) {
  return a_frag_from_row_f32((const float4*)ldsrow, lane);
}

// D fragment -> global f32 [nRows x 32] row-major with row guard.
__device__ __forceinline__ void store_d_f32(float* out, int base, int nRows,
                                            v8f d, int n0, int lane) {
  const int n  = n0 + (lane & 15);
  const int mb = (lane & 16) ? 8 : 0;
  #pragma unroll
  for (int r = 0; r < 8; ++r) {
    const int m = base + mb + r;
    if (m < nRows) out[(size_t)m * 32 + n] = d[r];
  }
}

// D fragment -> global f16 [nRows x 32] row-major with row guard.
__device__ __forceinline__ void store_d_f16(_Float16* out, int base, int nRows,
                                            v8f d, int n0, int lane) {
  const int n  = n0 + (lane & 15);
  const int mb = (lane & 16) ? 8 : 0;
  #pragma unroll
  for (int r = 0; r < 8; ++r) {
    const int m = base + mb + r;
    if (m < nRows) out[(size_t)m * 32 + n] = (_Float16)d[r];
  }
}

__device__ __forceinline__ void atomic_add_f32(float* p, float v) {
  unsafeAtomicAdd(p, v);   // native global_atomic_add_f32 (L2 atomic units)
}

// ---------------------------------------------------------------------------
// Encoder: x = relu(nf[64] @ w1) @ w2 + biases -> f16 x
// ---------------------------------------------------------------------------
__global__ void __launch_bounds__(TPB)
gnn_encoder(const float* __restrict__ nf,
            const float* __restrict__ w1, const float* __restrict__ b1,
            const float* __restrict__ w2, const float* __restrict__ b2,
            _Float16* __restrict__ xout, int nNodes) {
  __shared__ __align__(16) float lds[WPB][16 * 32];
  const int lane = threadIdx.x & 31;
  const int wv   = threadIdx.x >> 5;

  v16h W1[2][2], W2[2];
  #pragma unroll
  for (int kt = 0; kt < 2; ++kt) {
    W1[kt][0] = b_frag_from_w(w1, 64, kt * 32, 0,  lane);
    W1[kt][1] = b_frag_from_w(w1, 64, kt * 32, 16, lane);
  }
  W2[0] = b_frag_from_w(w2, 32, 0, 0,  lane);
  W2[1] = b_frag_from_w(w2, 32, 0, 16, lane);
  const float b1lo = b1[lane & 15], b1hi = b1[16 + (lane & 15)];
  const float b2lo = b2[lane & 15], b2hi = b2[16 + (lane & 15)];

  const int nTiles = (nNodes + 15) >> 4;
  for (int t = blockIdx.x * WPB + wv; t < nTiles; t += gridDim.x * WPB) {
    const int base = t * 16;
    int node = base + (lane & 15);
    if (node >= nNodes) node = nNodes - 1;
    const float4* row4 = (const float4*)(nf + (size_t)node * 64);

    v16h a0 = a_frag_from_row_f32(row4,     lane);   // K 0..31
    v16h a1 = a_frag_from_row_f32(row4 + 8, lane);   // K 32..63

    v8f c0 = splat8(b1lo), c1 = splat8(b1hi);
    c0 = wmma16(a0, W1[0][0], c0);  c1 = wmma16(a0, W1[0][1], c1);
    c0 = wmma16(a1, W1[1][0], c0);  c1 = wmma16(a1, W1[1][1], c1);

    relu_store_lds(lds[wv], c0, 0,  lane);
    relu_store_lds(lds[wv], c1, 16, lane);
    v16h h = a_frag_from_lds(lds[wv] + (lane & 15) * 32, lane);

    v8f d0 = splat8(b2lo), d1 = splat8(b2hi);
    d0 = wmma16(h, W2[0], d0);
    d1 = wmma16(h, W2[1], d1);

    store_d_f16(xout, base, nNodes, d0, 0,  lane);
    store_d_f16(xout, base, nNodes, d1, 16, lane);
  }
}

// ---------------------------------------------------------------------------
// Edge message: msg = mlp2([x[dst], x[src], dist]); agg[dst] += msg (atomics)
// K padded 65 -> 96 (tile2 carries only dist at k=64).  x gathers are f16.
// ---------------------------------------------------------------------------
__global__ void __launch_bounds__(TPB)
gnn_edge_msg(const _Float16* __restrict__ x, const int* __restrict__ ei,
             const float* __restrict__ pos,
             const float* __restrict__ w1, const float* __restrict__ b1,
             const float* __restrict__ w2, const float* __restrict__ b2,
             float* __restrict__ agg, int nEdges) {
  __shared__ __align__(16) float lds[WPB][16 * 32];
  __shared__ int ldsDst[WPB][16];
  const int lane = threadIdx.x & 31;
  const int wv   = threadIdx.x >> 5;

  v16h W1[3][2], W2[2];
  #pragma unroll
  for (int kt = 0; kt < 3; ++kt) {
    W1[kt][0] = b_frag_from_w(w1, 65, kt * 32, 0,  lane);
    W1[kt][1] = b_frag_from_w(w1, 65, kt * 32, 16, lane);
  }
  W2[0] = b_frag_from_w(w2, 32, 0, 0,  lane);
  W2[1] = b_frag_from_w(w2, 32, 0, 16, lane);
  const float b1lo = b1[lane & 15], b1hi = b1[16 + (lane & 15)];
  const float b2lo = b2[lane & 15], b2hi = b2[16 + (lane & 15)];

  const int nTiles = (nEdges + 15) >> 4;
  for (int t = blockIdx.x * WPB + wv; t < nTiles; t += gridDim.x * WPB) {
    const int base = t * 16;
    const int m = lane & 15;
    int e = base + m;
    if (e >= nEdges) e = nEdges - 1;
    const int src = ei[e];
    const int dst = ei[nEdges + e];
    ldsDst[wv][m] = dst;                       // both half-waves write same value

    const float dx = pos[dst * 3 + 0] - pos[src * 3 + 0];
    const float dy = pos[dst * 3 + 1] - pos[src * 3 + 1];
    const float dz = pos[dst * 3 + 2] - pos[src * 3 + 2];
    const float dist = sqrtf(dx * dx + dy * dy + dz * dz);

    v16h a0 = a_frag_from_row_f16(x + (size_t)dst * 32, lane);  // K 0..31
    v16h a1 = a_frag_from_row_f16(x + (size_t)src * 32, lane);  // K 32..63
    v16h a2;                                                    // K 64..95
    #pragma unroll
    for (int h = 0; h < 16; ++h) a2[h] = (_Float16)0.0f;
    a2[0] = (lane & 16) ? (_Float16)0.0f : (_Float16)dist;      // k=64

    v8f c0 = splat8(b1lo), c1 = splat8(b1hi);
    c0 = wmma16(a0, W1[0][0], c0);  c1 = wmma16(a0, W1[0][1], c1);
    c0 = wmma16(a1, W1[1][0], c0);  c1 = wmma16(a1, W1[1][1], c1);
    c0 = wmma16(a2, W1[2][0], c0);  c1 = wmma16(a2, W1[2][1], c1);

    relu_store_lds(lds[wv], c0, 0,  lane);
    relu_store_lds(lds[wv], c1, 16, lane);
    v16h h1 = a_frag_from_lds(lds[wv] + m * 32, lane);

    v8f d0 = splat8(b2lo), d1 = splat8(b2hi);
    d0 = wmma16(h1, W2[0], d0);
    d1 = wmma16(h1, W2[1], d1);

    const int n  = lane & 15;
    const int mb = (lane & 16) ? 8 : 0;
    #pragma unroll
    for (int r = 0; r < 8; ++r) {
      if (base + mb + r < nEdges) {
        float* p = agg + (size_t)ldsDst[wv][mb + r] * 32;
        atomic_add_f32(p + n,      d0[r]);
        atomic_add_f32(p + 16 + n, d1[r]);
      }
    }
  }
}

// ---------------------------------------------------------------------------
// Update: x' = relu([x, agg] @ w1) @ w2 + biases -> f16 x'
// ---------------------------------------------------------------------------
__global__ void __launch_bounds__(TPB)
gnn_update(const _Float16* __restrict__ x, const float* __restrict__ agg,
           const float* __restrict__ w1, const float* __restrict__ b1,
           const float* __restrict__ w2, const float* __restrict__ b2,
           _Float16* __restrict__ xout, int nNodes) {
  __shared__ __align__(16) float lds[WPB][16 * 32];
  const int lane = threadIdx.x & 31;
  const int wv   = threadIdx.x >> 5;

  v16h W1[2][2], W2[2];
  #pragma unroll
  for (int kt = 0; kt < 2; ++kt) {
    W1[kt][0] = b_frag_from_w(w1, 64, kt * 32, 0,  lane);
    W1[kt][1] = b_frag_from_w(w1, 64, kt * 32, 16, lane);
  }
  W2[0] = b_frag_from_w(w2, 32, 0, 0,  lane);
  W2[1] = b_frag_from_w(w2, 32, 0, 16, lane);
  const float b1lo = b1[lane & 15], b1hi = b1[16 + (lane & 15)];
  const float b2lo = b2[lane & 15], b2hi = b2[16 + (lane & 15)];

  const int nTiles = (nNodes + 15) >> 4;
  for (int t = blockIdx.x * WPB + wv; t < nTiles; t += gridDim.x * WPB) {
    const int base = t * 16;
    int node = base + (lane & 15);
    if (node >= nNodes) node = nNodes - 1;

    v16h a0 = a_frag_from_row_f16(x + (size_t)node * 32, lane);
    v16h a1 = a_frag_from_row_f32((const float4*)(agg + (size_t)node * 32), lane);

    v8f c0 = splat8(b1lo), c1 = splat8(b1hi);
    c0 = wmma16(a0, W1[0][0], c0);  c1 = wmma16(a0, W1[0][1], c1);
    c0 = wmma16(a1, W1[1][0], c0);  c1 = wmma16(a1, W1[1][1], c1);

    relu_store_lds(lds[wv], c0, 0,  lane);
    relu_store_lds(lds[wv], c1, 16, lane);
    v16h h = a_frag_from_lds(lds[wv] + (lane & 15) * 32, lane);

    v8f d0 = splat8(b2lo), d1 = splat8(b2hi);
    d0 = wmma16(h, W2[0], d0);
    d1 = wmma16(h, W2[1], d1);

    store_d_f16(xout, base, nNodes, d0, 0,  lane);
    store_d_f16(xout, base, nNodes, d1, 16, lane);
  }
}

// ---------------------------------------------------------------------------
// Output head: out = x @ out_w + out_b   (f32 output)
// ---------------------------------------------------------------------------
__global__ void __launch_bounds__(TPB)
gnn_out(const _Float16* __restrict__ x,
        const float* __restrict__ w, const float* __restrict__ b,
        float* __restrict__ out, int nNodes) {
  const int lane = threadIdx.x & 31;
  const int wv   = threadIdx.x >> 5;

  v16h W[2];
  W[0] = b_frag_from_w(w, 32, 0, 0,  lane);
  W[1] = b_frag_from_w(w, 32, 0, 16, lane);
  const float blo = b[lane & 15], bhi = b[16 + (lane & 15)];

  const int nTiles = (nNodes + 15) >> 4;
  for (int t = blockIdx.x * WPB + wv; t < nTiles; t += gridDim.x * WPB) {
    const int base = t * 16;
    int node = base + (lane & 15);
    if (node >= nNodes) node = nNodes - 1;

    v16h a  = a_frag_from_row_f16(x + (size_t)node * 32, lane);
    v8f  d0 = splat8(blo), d1 = splat8(bhi);
    d0 = wmma16(a, W[0], d0);
    d1 = wmma16(a, W[1], d1);

    store_d_f32(out, base, nNodes, d0, 0,  lane);
    store_d_f32(out, base, nNodes, d1, 16, lane);
  }
}

// ---------------------------------------------------------------------------
// Host driver
// ---------------------------------------------------------------------------
extern "C" void kernel_launch(void* const* d_in, const int* in_sizes, int n_in,
                              void* d_out, int out_size, void* d_ws, size_t ws_size,
                              hipStream_t stream) {
  const float* node_feat = (const float*)d_in[0];
  const float* pos       = (const float*)d_in[1];
  const int*   ei        = (const int*)  d_in[2];
  const float* enc_w1 = (const float*)d_in[3];
  const float* enc_b1 = (const float*)d_in[4];
  const float* enc_w2 = (const float*)d_in[5];
  const float* enc_b2 = (const float*)d_in[6];
  const float* msg_w1 = (const float*)d_in[7];
  const float* msg_b1 = (const float*)d_in[8];
  const float* msg_w2 = (const float*)d_in[9];
  const float* msg_b2 = (const float*)d_in[10];
  const float* upd_w1 = (const float*)d_in[11];
  const float* upd_b1 = (const float*)d_in[12];
  const float* upd_w2 = (const float*)d_in[13];
  const float* upd_b2 = (const float*)d_in[14];
  const float* out_w  = (const float*)d_in[15];
  const float* out_b  = (const float*)d_in[16];

  const int nNodes = in_sizes[0] / 64;   // node_feat is [N, 64]
  const int nEdges = in_sizes[2] / 2;    // edge_index is [2, E]
  const int NUM_LAYERS = 3;

  // Workspace: agg (f32, 16B-aligned first) | x0 (f16) | x1 (f16).
  float*    agg = (float*)d_ws;
  _Float16* x0  = (_Float16*)(agg + (size_t)nNodes * 32);
  _Float16* x1  = x0 + (size_t)nNodes * 32;

  const int nodeTiles = (nNodes + 15) / 16;
  const int edgeTiles = (nEdges + 15) / 16;
  int nodeBlocks = (nodeTiles + WPB - 1) / WPB;
  if (nodeBlocks > 512)  nodeBlocks = 512;
  int edgeBlocks = (edgeTiles + WPB - 1) / WPB;
  if (edgeBlocks > 2048) edgeBlocks = 2048;

  gnn_encoder<<<nodeBlocks, TPB, 0, stream>>>(node_feat, enc_w1, enc_b1,
                                              enc_w2, enc_b2, x0, nNodes);

  _Float16* xin  = x0;
  _Float16* xout = x1;
  for (int l = 0; l < NUM_LAYERS; ++l) {
    hipMemsetAsync(agg, 0, (size_t)nNodes * 32 * sizeof(float), stream);
    gnn_edge_msg<<<edgeBlocks, TPB, 0, stream>>>(
        xin, ei, pos,
        msg_w1 + (size_t)l * 65 * 32, msg_b1 + (size_t)l * 32,
        msg_w2 + (size_t)l * 32 * 32, msg_b2 + (size_t)l * 32,
        agg, nEdges);
    gnn_update<<<nodeBlocks, TPB, 0, stream>>>(
        xin, agg,
        upd_w1 + (size_t)l * 64 * 32, upd_b1 + (size_t)l * 32,
        upd_w2 + (size_t)l * 32 * 32, upd_b2 + (size_t)l * 32,
        xout, nNodes);
    _Float16* tmp = xin; xin = xout; xout = tmp;
  }

  gnn_out<<<nodeBlocks, TPB, 0, stream>>>(xin, out_w, out_b, (float*)d_out, nNodes);
}
